// Model_46952582480187
// MI455X (gfx1250) — matrix operand order repaired and checked by reading, not compiled, not run on hardware
//
#include <hip/hip_runtime.h>

// ---------------------------------------------------------------------------
// LSTM (relu candidate/output, sigmoid gates) + dense head, bf16 WMMA path.
//   B=256, T=512, F=32, U=350.  4U=1400 -> padded N=1408 (88 tiles of 16).
//   U=350 -> padded K=352 (11 blocks of 32).  One WG per 16 batch rows.
// Recurrent weights (bf16, pre-packed into WMMA B-fragment layout) are
// streamed per K-block into LDS with the Tensor Data Mover, double-buffered
// per wave, synchronized with s_wait_tensorcnt.
// ---------------------------------------------------------------------------

#define T_    512
#define F_    32
#define U_    350
#define G4    1400
#define NP    1408
#define NT    88          // N tiles of 16
#define KB    11          // K blocks of 32 for recurrent matmul
#define ZROW  1409        // LDS z row stride (floats) — stagger banks
#define HROW  360         // LDS h row stride (bf16)   — 16B aligned, staggered
#define WCHUNK (11 * 512) // bf16 elements per wave per K-block (11,264 B)

typedef __bf16 v16bf __attribute__((ext_vector_type(16)));
typedef float  v8f   __attribute__((ext_vector_type(8)));
typedef unsigned int u32x4 __attribute__((ext_vector_type(4)));
typedef int          i32x4 __attribute__((ext_vector_type(4)));
typedef int          i32x8 __attribute__((ext_vector_type(8)));

union FragAB { v16bf v; uint4 q[2]; };

__device__ __forceinline__ unsigned short f2bf(float f) {
  unsigned int u = __float_as_uint(f);
  u += 0x7FFFu + ((u >> 16) & 1u);      // round-to-nearest-even
  return (unsigned short)(u >> 16);
}
__device__ __forceinline__ float sigm(float x) {
  return 1.0f / (1.0f + __expf(-x));
}

// TDM: DMA one contiguous 11,264-byte chunk (1408 x 8B elements, 1-row tile)
// from global memory into LDS.  D# per CDNA5 ISA §8.3/8.4.
__device__ __forceinline__ void tdm_load_chunk(const void* gsrc,
                                               unsigned int lds_addr) {
  unsigned long long ga = (unsigned long long)gsrc;
  u32x4 g0 = { 1u,                       // [1:0] count=1, user load
               lds_addr,                 // [63:32] LDS byte address
               (unsigned int)ga,         // [95:64] global addr lo
               (unsigned int)((ga >> 32) & 0x01FFFFFFu) | (2u << 30) }; // type=2
  i32x8 g1 = { (int)0x00030000,          // data_size=3 (8B), wg_mask=0
               (int)(1408u << 16),       // tensor_dim0[15:0] @ bits[63:48]
               (int)(1u << 16),          // tensor_dim0 hi=0 | tensor_dim1=1
               (int)(1408u << 16),       // tile_dim0=1408 @ bits[127:112]
               1,                        // tile_dim1=1, tile_dim2=0
               1408,                     // tensor_dim0_stride
               0, 0 };
  i32x4 gz = { 0, 0, 0, 0 };
#if __clang_major__ >= 23
  i32x8 gz8 = { 0, 0, 0, 0, 0, 0, 0, 0 };
  __builtin_amdgcn_tensor_load_to_lds(g0, g1, gz, gz, gz8, 0);
#else
  __builtin_amdgcn_tensor_load_to_lds(g0, g1, gz, gz, 0);
#endif
}

// ---------------------------------------------------------------------------
// Pre-pass kernels: convert / pack into WMMA fragment layouts.
// B-fragment layout used here (contiguous k, not the ISA table order): lane L holds column n = tile*16 + (L&15);
// half-slot s in 0..15 holds K = kblock*32 + (L<16 ? s : 16+s).
// ---------------------------------------------------------------------------

__global__ void lstm46_pack_xbf(const float* __restrict__ x,
                                unsigned short* __restrict__ xbf, int n) {
  for (int i = blockIdx.x * blockDim.x + threadIdx.x; i < n;
       i += gridDim.x * blockDim.x)
    xbf[i] = f2bf(x[i]);
}

__global__ void lstm46_pack_k(const float* __restrict__ K,   // [32 x 1400]
                              unsigned short* __restrict__ pk) {
  int idx = blockIdx.x * blockDim.x + threadIdx.x;           // NT*512 total
  if (idx >= NT * 512) return;
  int nb = idx >> 9, r = idx & 511, lane = r >> 4, s = r & 15;
  int n = nb * 16 + (lane & 15);
  int k = (lane < 16) ? s : 16 + s;                          // F=32 == one block
  pk[idx] = f2bf((n < G4) ? K[k * G4 + n] : 0.0f);
}

__global__ void lstm46_pack_w(const float* __restrict__ W,   // [350 x 1400]
                              unsigned short* __restrict__ pw) {
  int idx = blockIdx.x * blockDim.x + threadIdx.x;           // KB*NT*512 total
  if (idx >= KB * NT * 512) return;
  int kb = idx / (NT * 512), rem = idx % (NT * 512);
  int nb = rem >> 9, r = rem & 511, lane = r >> 4, s = r & 15;
  int k = kb * 32 + ((lane < 16) ? s : 16 + s);
  int n = nb * 16 + (lane & 15);
  pw[idx] = f2bf((k < U_ && n < G4) ? W[k * G4 + n] : 0.0f);
}

// ---------------------------------------------------------------------------
// Main persistent-per-batch-tile LSTM kernel. 16 blocks x 256 threads (8 waves).
// Wave w owns N-tiles [w*11, w*11+11).
// ---------------------------------------------------------------------------

__global__ __launch_bounds__(256)
void lstm46_main(const unsigned short* __restrict__ xbf,  // [B,T,32] bf16
                 const unsigned short* __restrict__ pk,   // packed kernel frags
                 const unsigned short* __restrict__ pw,   // packed recurrent frags
                 const float* __restrict__ bias,          // [1400]
                 const float* __restrict__ dw,            // [350]
                 const float* __restrict__ dbp,           // [1]
                 float* __restrict__ out) {               // [B,T]
  extern __shared__ char smem_raw[];
  float*          zS    = (float*)smem_raw;               // 16*ZROW f32
  float*          cS    = zS + 16 * ZROW;                 // 16*350  f32
  float*          biasS = cS + 16 * U_;                   // 1408    f32
  float*          dwS   = biasS + NP;                     // 352     f32
  unsigned short* hS    = (unsigned short*)(dwS + 352);   // 16*HROW bf16
  unsigned short* wbS   = hS + 16 * HROW;                 // 2*8*WCHUNK bf16

  const int tid  = threadIdx.x;
  const int lane = tid & 31;
  const int wave = tid >> 5;
  const int m16  = lane & 15;
  const int hsel = lane >> 4;          // 0: lanes 0-15, 1: lanes 16-31
  const int k0   = hsel * 16;          // A-frag: lane half h holds K = 16h + s (matches packed B)
  const int b0   = blockIdx.x * 16;
  const float db = dbp[0];

  for (int i = tid; i < 16 * U_;  i += 256) cS[i] = 0.0f;
  for (int i = tid; i < 16 * HROW; i += 256) hS[i] = 0;
  for (int i = tid; i < NP;  i += 256) biasS[i] = (i < G4) ? bias[i] : 0.0f;
  for (int i = tid; i < 352; i += 256) dwS[i]   = (i < U_) ? dw[i]   : 0.0f;
  __syncthreads();

  // per-wave TDM double buffers (LDS byte addresses) and global chunk base
  const unsigned int wb_addr[2] = {
    (unsigned int)(unsigned long long)(wbS + (0 * 8 + wave) * WCHUNK),
    (unsigned int)(unsigned long long)(wbS + (1 * 8 + wave) * WCHUNK) };
  const unsigned short* wchunk = pw + wave * WCHUNK;   // + kb*NT*512 per block

  int buf = 0;
  tdm_load_chunk(wchunk, wb_addr[0]);                  // prime kb=0 of t=0

  for (int t = 0; t < T_; ++t) {
    // ---- z = bias ---------------------------------------------------------
    v8f acc[11];
#pragma unroll
    for (int j = 0; j < 11; ++j) {
      float bv = biasS[(wave * 11 + j) * 16 + m16];
#pragma unroll
      for (int r = 0; r < 8; ++r) acc[j][r] = bv;
    }

    // ---- z += x_t @ kernel  (single K-block of 32 == F) -------------------
    {
      FragAB a;
      const unsigned short* xp =
          xbf + ((size_t)(b0 + m16) * T_ + t) * F_;
      a.q[0] = *(const uint4*)(xp + k0);
      a.q[1] = *(const uint4*)(xp + k0 + 8);
      if (t + 1 < T_)
        __builtin_prefetch(xp + F_ + k0, 0, 0);        // next step's x frag
#pragma unroll
      for (int j = 0; j < 11; ++j) {
        FragAB b;
        const unsigned short* bp = pk + ((wave * 11 + j) * 512) + lane * 16;
        b.q[0] = *(const uint4*)(bp);
        b.q[1] = *(const uint4*)(bp + 8);
        acc[j] = __builtin_amdgcn_wmma_f32_16x16x32_bf16(
            false, a.v, false, b.v, (short)0, acc[j], false, false);
      }
    }

    // ---- z += h @ recurrent_kernel  (11 K-blocks, TDM double-buffered) ----
    for (int kb = 0; kb < KB; ++kb) {
      // DMA next K-block (wraps to kb=0, priming the next timestep)
      int nkb = (kb + 1 == KB) ? 0 : kb + 1;
      tdm_load_chunk(wchunk + (size_t)nkb * (NT * 512), wb_addr[buf ^ 1]);
      __builtin_amdgcn_s_wait_tensorcnt(1);  // current buffer's DMA complete

      FragAB a;
      const unsigned short* hp = hS + m16 * HROW + kb * 32 + k0;
      a.q[0] = *(const uint4*)(hp);
      a.q[1] = *(const uint4*)(hp + 8);
      const unsigned short* wb = wbS + (buf * 8 + wave) * WCHUNK;
#pragma unroll
      for (int j = 0; j < 11; ++j) {
        FragAB b;
        const unsigned short* bp = wb + j * 512 + lane * 16;
        b.q[0] = *(const uint4*)(bp);
        b.q[1] = *(const uint4*)(bp + 8);
        acc[j] = __builtin_amdgcn_wmma_f32_16x16x32_bf16(
            false, a.v, false, b.v, (short)0, acc[j], false, false);
      }
      buf ^= 1;
    }

    // ---- scatter z to LDS (C layout: reg r -> row r + 8*hsel) -------------
#pragma unroll
    for (int j = 0; j < 11; ++j) {
      int col = (wave * 11 + j) * 16 + m16;
#pragma unroll
      for (int r = 0; r < 8; ++r)
        zS[(r + hsel * 8) * ZROW + col] = acc[j][r];
    }
    __syncthreads();

    // ---- gates + state update + folded dense head -------------------------
    const int m  = tid >> 4;   // batch row in tile
    const int ul = tid & 15;   // unit lane
    float dsum = 0.0f;
    for (int u = ul; u < U_; u += 16) {
      float zi = zS[m * ZROW + u];
      float zf = zS[m * ZROW + U_ + u];
      float zg = zS[m * ZROW + 2 * U_ + u];
      float zo = zS[m * ZROW + 3 * U_ + u];
      float ig = sigm(zi);
      float fg = sigm(zf);
      float g  = fmaxf(zg, 0.0f);            // activation='relu' on candidate
      float og = sigm(zo);
      float cn = fg * cS[m * U_ + u] + ig * g;
      float hn = og * fmaxf(cn, 0.0f);       // activation='relu' on output
      cS[m * U_ + u] = cn;
      hS[m * HROW + u] = f2bf(hn);
      dsum += hn * dwS[u];
    }
#pragma unroll
    for (int off = 8; off; off >>= 1) dsum += __shfl_xor(dsum, off, 16);
    if (ul == 0) out[(size_t)(b0 + m) * T_ + t] = dsum + db;
    __syncthreads();   // h/c ready before next step's A-frag reads
  }
}

// ---------------------------------------------------------------------------

extern "C" void kernel_launch(void* const* d_in, const int* in_sizes, int n_in,
                              void* d_out, int out_size, void* d_ws, size_t ws_size,
                              hipStream_t stream) {
  const float* x    = (const float*)d_in[0];   // [256,512,32]
  const float* kern = (const float*)d_in[1];   // [32,1400]
  const float* rec  = (const float*)d_in[2];   // [350,1400]
  const float* bias = (const float*)d_in[3];   // [1400]
  const float* dw   = (const float*)d_in[4];   // [350,1]
  const float* db   = (const float*)d_in[5];   // [1]
  float* out = (float*)d_out;                  // [256,512,1]

  // workspace carve (≈9.5 MB total)
  const size_t XBF_BYTES = (size_t)256 * T_ * F_ * 2;       // 8,388,608
  const size_t PK_BYTES  = (size_t)NT * 512 * 2;            //    90,112
  unsigned short* xbf = (unsigned short*)d_ws;
  unsigned short* pk  = (unsigned short*)((char*)d_ws + XBF_BYTES);
  unsigned short* pw  = (unsigned short*)((char*)d_ws + XBF_BYTES + PK_BYTES);

  const int nX = 256 * T_ * F_;
  lstm46_pack_xbf<<<(nX + 255) / 256, 256, 0, stream>>>(x, xbf, nX);
  lstm46_pack_k  <<<(NT * 512 + 255) / 256, 256, 0, stream>>>(kern, pk);
  lstm46_pack_w  <<<(KB * NT * 512 + 255) / 256, 256, 0, stream>>>(rec, pw);

  const int smemBytes =
      (16 * ZROW + 16 * U_ + NP + 352) * 4 + 16 * HROW * 2
      + 2 * 8 * WCHUNK * 2;                                 // 311,360 B
  (void)hipFuncSetAttribute((const void*)lstm46_main,
                            hipFuncAttributeMaxDynamicSharedMemorySize,
                            smemBytes);
  lstm46_main<<<16, 256, smemBytes, stream>>>(xbf, pk, pw, bias, dw, db, out);
}
